// ResNet_6674379178404
// MI455X (gfx1250) — matrix operand order repaired and checked by reading, not compiled, not run on hardware
//
#include <hip/hip_runtime.h>
#include <hip/hip_bf16.h>

// ---------------------------------------------------------------------------
// CDNA5 (gfx1250): implicit-GEMM conv, v_wmma_f32_16x16x32_f16 + TDM staging.
// Block = 128 threads (4 waves). Block tile = 64(M) x 32(N), K-step = 64.
// Each wave: 16(M) x 32(N) -> 2 f32 accumulators, 4 WMMAs per K stage.
// A (weight) tile staged by TENSOR_LOAD_TO_LDS (f16, row-padded by descriptor);
// B (im2col) tile staged by the 4 waves while the TDM runs.
// ---------------------------------------------------------------------------

typedef __attribute__((ext_vector_type(16))) _Float16 v16h;
typedef __attribute__((ext_vector_type(8)))  float    v8f;
typedef __attribute__((ext_vector_type(4)))  float    v4f;
typedef __attribute__((ext_vector_type(4)))  unsigned int u32x4;
typedef __attribute__((ext_vector_type(8)))  int      i32x8;
typedef __attribute__((ext_vector_type(4)))  int      i32x4;

#define POOL_H 4
#define POOL_W 280

// LDS plan (dynamic segment, offset 0):
//   As: 64 rows x 72 halves  (row = 128B data + 16B pad)   bytes [0, 9216)
//   Bs: 32 rows x 72 halves                                 bytes [9216, 13824)
#define LDS_AROW   72
#define LDS_BOFF   (64 * 72)
#define LDS_BYTES  ((64 * 72 + 32 * 72) * 2)

#if defined(__has_builtin)
#  if __has_builtin(__builtin_amdgcn_tensor_load_to_lds) && \
      __has_builtin(__builtin_amdgcn_s_wait_tensorcnt)
#    define USE_TDM 1
#  endif
#endif
#ifndef USE_TDM
#  define USE_TDM 0
#endif

union FragU { v16h h; v4f f[2]; };

#if USE_TDM
// Issue one 2D TDM load: tile 64(rows) x 64(halves) from wgt16[mBase..][k0..],
// row stride K halves, into LDS offset 0 with 16B pad after each 128B row.
// tensor_dim0/1 = remaining extents -> hardware zero-fills the K/M tails.
__device__ __forceinline__ void tdm_load_wtile(const _Float16* wgt, long mBase,
                                               int k0, int K, int Cout)
{
    unsigned long long ga = (unsigned long long)(wgt + mBase * (long)K + k0);
    unsigned td0 = (unsigned)(K - k0);            // halves remaining in K
    unsigned td1 = (unsigned)(Cout - (int)mBase); // rows remaining in M

    u32x4 g0 = { 1u,                                   // count=1 (valid D#)
                 0u,                                   // lds_addr = 0
                 (unsigned)ga,                         // global_addr[31:0]
                 (unsigned)(ga >> 32) | (2u << 30) };  // addr[56:32] | type=2

    int w0 = (int)((1u << 16)        // data_size = 2 bytes
                 | (1u << 20)        // pad_enable
                 | (4u << 22)        // pad_interval: 32 DWORDs (128B)
                 | (3u << 25));      // pad_amount:   4 DWORDs (16B)
    i32x8 g1 = { w0,
                 (int)((td0 & 0xFFFFu) << 16),                       // dim0 lo
                 (int)((td0 >> 16) | ((td1 & 0xFFFFu) << 16)),       // dim0 hi|dim1 lo
                 (int)((td1 >> 16) | (64u << 16)),                   // dim1 hi|tile0=64
                 (int)64,                                            // tile1=64, tile2=0
                 (int)(unsigned)K,                                   // stride0 lo
                 0, 0 };                                             // stride0 hi, stride1
    i32x4 gz4 = { 0, 0, 0, 0 };
    i32x8 gz8 = { 0, 0, 0, 0, 0, 0, 0, 0 };
    __builtin_amdgcn_tensor_load_to_lds(g0, g1, gz4, gz4, gz8, 0);
}
#endif

// ---------------------------------------------------------------------------
// Generic fused conv kernel.
//   D[M,N] = sum_K W[M,K] * X[K,N];  M=Cout, N=Hout*Wout, K=Cin*KH*KW.
//   Epilogue: optional scale/bias (BN or conv bias), residual add, ReLU.
// ---------------------------------------------------------------------------
__global__ __launch_bounds__(128)
void conv_wmma_kernel(const float* __restrict__ in, const _Float16* __restrict__ wgt,
                      const float* __restrict__ scale, const float* __restrict__ bias,
                      const float* __restrict__ residual, float* __restrict__ out,
                      int Cin, int Hin, int Win, int Cout, int Hout, int Wout,
                      int KH, int KW, int stride, int padH, int padW, int relu,
                      long inBS, long outBS)
{
    extern __shared__ _Float16 lds[];   // dynamic: segment offset 0 (TDM target)

    const int tid   = threadIdx.x;
    const int lane  = tid & 31;
    const int wave  = tid >> 5;
    const int laneM = lane & 15;
    const int khalf = lane >> 4;

    const int N  = Hout * Wout;
    const int K  = Cin * KH * KW;
    const int KK = KH * KW;
    const int nBase = blockIdx.x * 32;
    const int mBase = blockIdx.y * 64;
    const int b     = blockIdx.z;
    const float* __restrict__ inB = in + (long)b * inBS;

    // per-thread im2col column: fixed n, walk k (loop-invariant part hoisted)
    const int nn = tid & 31;
    const int gn = nBase + nn;
    const bool nValid = gn < N;
    int ihBase = 0, iwBase = 0;
    if (nValid) {
        int oh = gn / Wout, ow = gn - oh * Wout;
        ihBase = oh * stride - padH;
        iwBase = ow * stride - padW;
    }

    v8f acc0 = {};
    v8f acc1 = {};

    for (int k0 = 0; k0 < K; k0 += 64) {
        if (k0 + 64 < K)
            __builtin_prefetch(wgt + (long)mBase * K + k0 + 64, 0, 1);

        // ---- A tile: 64(m) x 64(k) f16 weights -> LDS[0..) ----
#if USE_TDM
        if (wave == 0)
            tdm_load_wtile(wgt, mBase, k0, K, Cout);
#else
        for (int e = tid; e < 64 * 64; e += 128) {
            int kk = e & 63, m = e >> 6;
            int gm = mBase + m, gk = k0 + kk;
            _Float16 v = (_Float16)0.f;
            if (gm < Cout && gk < K) v = wgt[(long)gm * K + gk];
            lds[m * LDS_AROW + kk] = v;
        }
#endif
        // ---- B tile: 32(n) x 64(k) im2col -> LDS[BOFF..) (overlaps TDM) ----
        for (int kk = tid >> 5; kk < 64; kk += 4) {
            int gk = k0 + kk;
            float v = 0.f;
            if (nValid && gk < K) {
                int ci  = gk / KK;
                int rem = gk - ci * KK;
                int kh  = rem / KW, kw = rem - kh * KW;
                int ih  = ihBase + kh;
                int iw  = iwBase + kw;
                if (ih >= 0 && ih < Hin && iw >= 0 && iw < Win)
                    v = inB[((long)ci * Hin + ih) * Win + iw];
            }
            lds[LDS_BOFF + nn * LDS_AROW + kk] = (_Float16)v;
        }
#if USE_TDM
        if (wave == 0)
            __builtin_amdgcn_s_wait_tensorcnt(0);
#endif
        __syncthreads();

        // ---- 2 K-substeps x 2 N-fragments = 4 WMMAs, 16B LDS reads ----
        // A 16x32: lane->M; lanes 0-15 hold K 0..15, 16-31 K 16..31
        // B 32x16: lane->N; lanes 0-15 hold K 0..15, 16-31 K 16..31
        #pragma unroll
        for (int ks = 0; ks < 2; ++ks) {
            const int kb = ks * 32;
            FragU a, b0, b1;
            const _Float16* arow  = lds + (wave * 16 + laneM) * LDS_AROW;
            const _Float16* brow0 = lds + LDS_BOFF + laneM * LDS_AROW;
            const _Float16* brow1 = lds + LDS_BOFF + (16 + laneM) * LDS_AROW;
            a.f[0]  = *(const v4f*)(arow  + kb + khalf * 16);
            a.f[1]  = *(const v4f*)(arow  + kb + khalf * 16 + 8);
            b0.f[0] = *(const v4f*)(brow0 + kb + khalf * 16);
            b0.f[1] = *(const v4f*)(brow0 + kb + khalf * 16 + 8);
            b1.f[0] = *(const v4f*)(brow1 + kb + khalf * 16);
            b1.f[1] = *(const v4f*)(brow1 + kb + khalf * 16 + 8);
            acc0 = __builtin_amdgcn_wmma_f32_16x16x32_f16(false, a.h, false, b0.h,
                                                          (short)0, acc0, false, false);
            acc1 = __builtin_amdgcn_wmma_f32_16x16x32_f16(false, a.h, false, b1.h,
                                                          (short)0, acc1, false, false);
        }
        __syncthreads();
    }

    // ---- epilogue: C/D layout VGPR r -> M = r (lanes 0-15) / 8+r (lanes 16-31)
    float* __restrict__ outB = out + (long)b * outBS;
    #pragma unroll
    for (int j = 0; j < 2; ++j) {
        const v8f& acc = j ? acc1 : acc0;
        int n = nBase + j * 16 + laneM;
        #pragma unroll
        for (int r = 0; r < 8; ++r) {
            int m = mBase + wave * 16 + khalf * 8 + r;
            if (m < Cout && n < N) {
                float v = acc[r];
                if (scale)      v = v * scale[m] + (bias ? bias[m] : 0.f);
                else if (bias)  v += bias[m];
                if (residual)   v += residual[(long)b * outBS + (long)m * N + n];
                if (relu)       v = fmaxf(v, 0.f);
                outB[(long)m * N + n] = v;
            }
        }
    }
}

// ---------------------------------------------------------------------------
// Elementwise / pooling helpers
// ---------------------------------------------------------------------------
__global__ void cvt_f32_f16_kernel(const float* __restrict__ in, _Float16* __restrict__ out, long n)
{
    long i = (long)blockIdx.x * blockDim.x + threadIdx.x;
    if (i < n) out[i] = (_Float16)in[i];
}

__global__ void maxpool3x3s2_kernel(const float* __restrict__ in, float* __restrict__ out,
                                    int C, int Hin, int Win, int Hout, int Wout)
{
    long idx = (long)blockIdx.x * blockDim.x + threadIdx.x;
    long total = (long)C * Hout * Wout;
    if (idx >= total) return;
    int w = (int)(idx % Wout);
    long t = idx / Wout;
    int h = (int)(t % Hout);
    int c = (int)(t / Hout);
    float m = -__builtin_inff();
    #pragma unroll
    for (int i = 0; i < 3; ++i)
        #pragma unroll
        for (int j = 0; j < 3; ++j) {
            int ih = h * 2 - 1 + i, iw = w * 2 - 1 + j;
            if (ih >= 0 && ih < Hin && iw >= 0 && iw < Win)
                m = fmaxf(m, in[((long)c * Hin + ih) * Win + iw]);
        }
    out[idx] = m;
}

__global__ void add_up2_kernel(float* __restrict__ dst, const float* __restrict__ src,
                               int C, int H, int W)
{
    long idx = (long)blockIdx.x * blockDim.x + threadIdx.x;
    long total = (long)C * H * W;
    if (idx >= total) return;
    int w = (int)(idx % W);
    long t = idx / W;
    int h = (int)(t % H);
    int c = (int)(t / H);
    dst[idx] += src[((long)c * (H / 2) + (h / 2)) * (W / 2) + (w / 2)];
}

__global__ void relu_copy_kernel(const float* __restrict__ in, float* __restrict__ out, long n)
{
    long i = (long)blockIdx.x * blockDim.x + threadIdx.x;
    if (i < n) out[i] = fmaxf(in[i], 0.f);
}

__global__ void fill_zero_kernel(float* __restrict__ p, long n)
{
    long i = (long)blockIdx.x * blockDim.x + threadIdx.x;
    if (i < n) p[i] = 0.f;
}

__global__ void scatter_head_kernel(const float* __restrict__ o, float* __restrict__ dst,
                                    int C, int H, int W, int per, int sigmoid)
{
    long idx = (long)blockIdx.x * blockDim.x + threadIdx.x;
    long total = (long)C * H * W;
    if (idx >= total) return;
    long hw = idx % ((long)H * W);
    int  c  = (int)(idx / ((long)H * W));
    int  a  = c / per, p = c - a * per;
    float v = o[idx];
    if (sigmoid) v = 1.f / (1.f + __expf(-v));
    dst[(hw * (C / per) + a) * (long)per + p] = v;
}

__global__ void roi_pool_kernel(const float* __restrict__ feat, const float* __restrict__ rois,
                                float* __restrict__ out, int C, int H, int W)
{
    int box = blockIdx.x;
    int c   = blockIdx.y;
    const float scale = 0.125f;
    float x1 = rois[box * 4 + 0] * scale;
    float y1 = rois[box * 4 + 1] * scale;
    float x2 = rois[box * 4 + 2] * scale;
    float y2 = rois[box * 4 + 3] * scale;
    const float* __restrict__ f = feat + (long)c * H * W;
    float* __restrict__ o = out + ((long)box * C + c) * (POOL_H * POOL_W);
    for (int t = threadIdx.x; t < POOL_H * POOL_W; t += blockDim.x) {
        int ph = t / POOL_W, pw = t - ph * POOL_W;
        float ys = y1 + (ph + 0.5f) * (y2 - y1) / (float)POOL_H;
        float xs = x1 + (pw + 0.5f) * (x2 - x1) / (float)POOL_W;
        ys = fminf(fmaxf(ys, 0.f), (float)H - 1.f);
        xs = fminf(fmaxf(xs, 0.f), (float)W - 1.f);
        int y0 = (int)floorf(ys), x0 = (int)floorf(xs);
        int y1i = min(y0 + 1, H - 1), x1i = min(x0 + 1, W - 1);
        float wy = ys - (float)y0, wx = xs - (float)x0;
        float f00 = f[(long)y0 * W + x0],  f01 = f[(long)y0 * W + x1i];
        float f10 = f[(long)y1i * W + x0], f11 = f[(long)y1i * W + x1i];
        o[t] = f00 * (1.f - wy) * (1.f - wx) + f01 * (1.f - wy) * wx
             + f10 * wy * (1.f - wx)        + f11 * wy * wx;
    }
}

__global__ void rec_logsoftmax_kernel(const float* __restrict__ o, float* __restrict__ out,
                                      int B, int C, int Wd)
{
    long row = (long)blockIdx.x * blockDim.x + threadIdx.x;
    if (row >= (long)B * Wd) return;
    int b = (int)(row / Wd), w = (int)(row % Wd);
    const float* __restrict__ src = o + (long)b * C * Wd + w;
    float mx = -__builtin_inff();
    for (int c = 0; c < C; ++c) mx = fmaxf(mx, src[(long)c * Wd]);
    float s = 0.f;
    for (int c = 0; c < C; ++c) s += __expf(src[(long)c * Wd] - mx);
    float lg = logf(s);
    float* __restrict__ d = out + ((long)b * Wd + w) * C;
    for (int c = 0; c < C; ++c) d[c] = src[(long)c * Wd] - mx - lg;
}

__global__ void ner_mean_fc_kernel(const float* __restrict__ h, const float* __restrict__ w,
                                   const float* __restrict__ bias, float* __restrict__ out,
                                   int C, int Wd, int NC)
{
    __shared__ float mean[256];
    int bb = blockIdx.x;
    for (int c = threadIdx.x; c < C; c += blockDim.x) {
        const float* __restrict__ src = h + ((long)bb * C + c) * Wd;
        float s = 0.f;
        for (int t = 0; t < Wd; ++t) s += src[t];
        mean[c] = s / (float)Wd;
    }
    __syncthreads();
    for (int j = threadIdx.x; j < NC; j += blockDim.x) {
        float s = bias[j];
        for (int c = 0; c < C; ++c) s += mean[c] * w[(long)c * NC + j];
        out[(long)bb * NC + j] = s;
    }
}

// ---------------------------------------------------------------------------
// Host orchestration
// ---------------------------------------------------------------------------
static inline dim3 g1(long n) { return dim3((unsigned)((n + 255) / 256)); }

static void launch_conv(hipStream_t s, const float* in, const _Float16* w,
                        const float* scale, const float* bias, const float* res, float* out,
                        int Cin, int Hin, int Win, int Cout, int KH, int KW,
                        int stride, int padH, int padW, int relu,
                        int batch = 1, long inBS = 0, long outBS = 0)
{
    int Hout = (Hin + 2 * padH - KH) / stride + 1;
    int Wout = (Win + 2 * padW - KW) / stride + 1;
    int N = Hout * Wout;
    dim3 grid((N + 31) / 32, (Cout + 63) / 64, batch);
    conv_wmma_kernel<<<grid, 128, LDS_BYTES, s>>>(in, w, scale, bias, res, out,
        Cin, Hin, Win, Cout, Hout, Wout, KH, KW, stride, padH, padW, relu,
        inBS ? inBS : (long)Cin * Hin * Win,
        outBS ? outBS : (long)Cout * Hout * Wout);
}

struct Blk {
    const _Float16 *c1, *c2, *dw;
    const float *b1s, *b1b, *b2s, *b2b, *dbs, *dbb;
};

extern "C" void kernel_launch(void* const* d_in, const int* in_sizes, int n_in,
                              void* d_out, int out_size, void* d_ws, size_t ws_size,
                              hipStream_t stream)
{
    (void)in_sizes; (void)n_in; (void)out_size; (void)ws_size;

    // ---- workspace bump allocator ----
    char* wp = (char*)d_ws;
    auto A = [&](long nfloat) {
        float* r = (float*)wp;
        wp += (((long)nfloat * 4 + 255) / 256) * 256;
        return r;
    };
    auto A16 = [&](long nhalf) {
        _Float16* r = (_Float16*)wp;
        wp += (((long)nhalf * 2 + 255) / 256) * 256;
        return r;
    };

    // ---- parse inputs in setup_inputs() insertion order, converting weights ----
    int ii = 0;
    auto nxt = [&]() { return (const float*)d_in[ii++]; };
    auto cvt = [&](const float* w, long n) {
        _Float16* p = A16(n);
        cvt_f32_f16_kernel<<<g1(n), 256, 0, stream>>>(w, p, n);
        return (const _Float16*)p;
    };

    const float* img  = nxt();                       // (1,3,800,1024)
    const float* rois = nxt();                       // (64,4)
    const _Float16* w_conv1 = cvt(nxt(), 64L * 3 * 49);
    const float* bn1s = nxt();
    const float* bn1b = nxt();

    auto readBlk = [&](bool down, int inC, int planes) {
        Blk B{};
        B.c1  = cvt(nxt(), (long)planes * inC * 9);
        B.b1s = nxt(); B.b1b = nxt();
        B.c2  = cvt(nxt(), (long)planes * planes * 9);
        B.b2s = nxt(); B.b2b = nxt();
        if (down) {
            B.dw  = cvt(nxt(), (long)planes * inC);
            B.dbs = nxt(); B.dbb = nxt();
        }
        return B;
    };
    Blk L[4][2];
    L[0][0] = readBlk(false,  64,  64); L[0][1] = readBlk(false,  64,  64);
    L[1][0] = readBlk(true,   64, 128); L[1][1] = readBlk(false, 128, 128);
    L[2][0] = readBlk(true,  128, 256); L[2][1] = readBlk(false, 256, 256);
    L[3][0] = readBlk(true,  256, 512); L[3][1] = readBlk(false, 512, 512);

    const _Float16* p5_1 = cvt(nxt(), 256L * 512);
    const _Float16* p5_2 = cvt(nxt(), 256L * 256 * 9);
    const _Float16* p4_1 = cvt(nxt(), 256L * 256);
    const _Float16* p4_2 = cvt(nxt(), 256L * 256 * 9);
    const _Float16* p3_1 = cvt(nxt(), 256L * 128);
    const _Float16* p3_2 = cvt(nxt(), 256L * 256 * 9);
    const _Float16* p6w  = cvt(nxt(), 256L * 512 * 9);
    const _Float16* p7w  = cvt(nxt(), 256L * 256 * 9);

    const _Float16* regc[4];
    for (int k = 0; k < 4; ++k) regc[k] = cvt(nxt(), 256L * 256 * 9);
    const _Float16* reg_ow = cvt(nxt(), 36L * 256 * 9);
    const float* reg_ob = nxt();
    const _Float16* clsc[4];
    for (int k = 0; k < 4; ++k) clsc[k] = cvt(nxt(), 256L * 256 * 9);
    const _Float16* cls_ow = cvt(nxt(), 117L * 256 * 9);
    const float* cls_ob = nxt();
    const _Float16* rec_c1 = cvt(nxt(), 256L * 256 * POOL_H * 3);
    const _Float16* rec_ow = cvt(nxt(), 81L * 256);
    const float* rec_ob = nxt();
    const _Float16* ner_c1 = cvt(nxt(), 256L * 256 * POOL_H * 3);
    const float* ner_w = nxt();
    const float* ner_b = nxt();

    // ---- activations ----
    float* c1out = A(64L * 400 * 512);
    float* x0    = A(64L * 200 * 256);
    float* t64   = A(64L * 200 * 256);
    float* o64   = A(64L * 200 * 256);
    float* x1    = A(64L * 200 * 256);
    float* t128  = A(128L * 100 * 128);
    float* d128  = A(128L * 100 * 128);
    float* o128  = A(128L * 100 * 128);
    float* x2    = A(128L * 100 * 128);
    float* t256  = A(256L * 50 * 64);
    float* d256  = A(256L * 50 * 64);
    float* o256  = A(256L * 50 * 64);
    float* x3    = A(256L * 50 * 64);
    float* t512  = A(512L * 25 * 32);
    float* d512  = A(512L * 25 * 32);
    float* o512  = A(512L * 25 * 32);
    float* x4    = A(512L * 25 * 32);

    float* p5pre = A(256L * 25 * 32);
    float* p5    = A(256L * 25 * 32);
    float* p4pre = A(256L * 50 * 64);
    float* p4    = A(256L * 50 * 64);
    float* p3pre = A(256L * 100 * 128);
    float* p3    = A(256L * 100 * 128);
    float* p6    = A(256L * 13 * 16);
    float* p6r   = A(256L * 13 * 16);
    float* p7    = A(256L * 7 * 8);

    float* hA    = A(256L * 100 * 128);
    float* hB    = A(256L * 100 * 128);
    float* oHead = A(117L * 100 * 128);

    float* pooled = A(128L * 256 * POOL_H * POOL_W); // first 64 = real boxes
    float* rec_h  = A(64L * 256 * POOL_W);
    float* rec_o  = A(64L * 81 * POOL_W);
    float* ner_h  = A(128L * 256 * POOL_W);

    hipStream_t s = stream;

    // ---- stem ----
    launch_conv(s, img, w_conv1, bn1s, bn1b, nullptr, c1out, 3, 800, 1024, 64, 7, 7, 2, 3, 3, 1);
    maxpool3x3s2_kernel<<<g1(64L * 200 * 256), 256, 0, s>>>(c1out, x0, 64, 400, 512, 200, 256);

    // ---- residual blocks ----
    auto block_fw = [&](const float* x, const Blk& B, int stride, int Cin, int planes,
                        int H, int W, float* tmp, float* down, float* out) {
        int Ho = (H + 2 - 3) / stride + 1;
        int Wo = (W + 2 - 3) / stride + 1;
        launch_conv(s, x, B.c1, B.b1s, B.b1b, nullptr, tmp, Cin, H, W, planes, 3, 3, stride, 1, 1, 1);
        const float* res = x;
        if (B.dw) {
            launch_conv(s, x, B.dw, B.dbs, B.dbb, nullptr, down, Cin, H, W, planes, 1, 1, stride, 0, 0, 0);
            res = down;
        }
        launch_conv(s, tmp, B.c2, B.b2s, B.b2b, res, out, planes, Ho, Wo, planes, 3, 3, 1, 1, 1, 1);
    };

    block_fw(x0,   L[0][0], 1,  64,  64, 200, 256, t64,  nullptr, o64);
    block_fw(o64,  L[0][1], 1,  64,  64, 200, 256, t64,  nullptr, x1);
    block_fw(x1,   L[1][0], 2,  64, 128, 200, 256, t128, d128,    o128);
    block_fw(o128, L[1][1], 1, 128, 128, 100, 128, t128, d128,    x2);
    block_fw(x2,   L[2][0], 2, 128, 256, 100, 128, t256, d256,    o256);
    block_fw(o256, L[2][1], 1, 256, 256,  50,  64, t256, d256,    x3);
    block_fw(x3,   L[3][0], 2, 256, 512,  50,  64, t512, d512,    o512);
    block_fw(o512, L[3][1], 1, 512, 512,  25,  32, t512, d512,    x4);

    // ---- FPN ----
    launch_conv(s, x4, p5_1, nullptr, nullptr, nullptr, p5pre, 512, 25, 32, 256, 1, 1, 1, 0, 0, 0);
    launch_conv(s, p5pre, p5_2, nullptr, nullptr, nullptr, p5, 256, 25, 32, 256, 3, 3, 1, 1, 1, 0);
    launch_conv(s, x3, p4_1, nullptr, nullptr, nullptr, p4pre, 256, 50, 64, 256, 1, 1, 1, 0, 0, 0);
    add_up2_kernel<<<g1(256L * 50 * 64), 256, 0, s>>>(p4pre, p5pre, 256, 50, 64);
    launch_conv(s, p4pre, p4_2, nullptr, nullptr, nullptr, p4, 256, 50, 64, 256, 3, 3, 1, 1, 1, 0);
    launch_conv(s, x2, p3_1, nullptr, nullptr, nullptr, p3pre, 128, 100, 128, 256, 1, 1, 1, 0, 0, 0);
    add_up2_kernel<<<g1(256L * 100 * 128), 256, 0, s>>>(p3pre, p4pre, 256, 100, 128);
    launch_conv(s, p3pre, p3_2, nullptr, nullptr, nullptr, p3, 256, 100, 128, 256, 3, 3, 1, 1, 1, 0);
    launch_conv(s, x4, p6w, nullptr, nullptr, nullptr, p6, 512, 25, 32, 256, 3, 3, 2, 1, 1, 0);
    relu_copy_kernel<<<g1(256L * 13 * 16), 256, 0, s>>>(p6, p6r, 256L * 13 * 16);
    launch_conv(s, p6r, p7w, nullptr, nullptr, nullptr, p7, 256, 13, 16, 256, 3, 3, 2, 1, 1, 0);

    // ---- detection heads ----
    const long TOTAL_HW = 12800 + 3200 + 800 + 208 + 56;       // 17064
    const long TOTAL_ANCH = TOTAL_HW * 9;                      // 153576
    float* cls_out   = (float*)d_out;
    float* reg_out   = cls_out + TOTAL_ANCH * 13;
    float* trans_out = reg_out + TOTAL_ANCH * 4;
    float* ner_out   = trans_out + 64L * POOL_W * 81;

    struct Feat { const float* f; int H, W; long off; };
    Feat feats[5] = {{p3, 100, 128, 0}, {p4, 50, 64, 12800}, {p5, 25, 32, 16000},
                     {p6, 13, 16, 16800}, {p7, 7, 8, 17008}};

    for (int i = 0; i < 5; ++i) {
        int H = feats[i].H, W = feats[i].W;
        long off = feats[i].off;
        // regression head
        const float* cur = feats[i].f;
        for (int k = 0; k < 4; ++k) {
            float* dst = (k & 1) ? hB : hA;
            launch_conv(s, cur, regc[k], nullptr, nullptr, nullptr, dst, 256, H, W, 256, 3, 3, 1, 1, 1, 1);
            cur = dst;
        }
        launch_conv(s, cur, reg_ow, nullptr, reg_ob, nullptr, oHead, 256, H, W, 36, 3, 3, 1, 1, 1, 0);
        scatter_head_kernel<<<g1(36L * H * W), 256, 0, s>>>(oHead, reg_out + off * 9 * 4, 36, H, W, 4, 0);
        // classification head (fused sigmoid on scatter)
        cur = feats[i].f;
        for (int k = 0; k < 4; ++k) {
            float* dst = (k & 1) ? hB : hA;
            launch_conv(s, cur, clsc[k], nullptr, nullptr, nullptr, dst, 256, H, W, 256, 3, 3, 1, 1, 1, 1);
            cur = dst;
        }
        launch_conv(s, cur, cls_ow, nullptr, cls_ob, nullptr, oHead, 256, H, W, 117, 3, 3, 1, 1, 1, 0);
        scatter_head_kernel<<<g1(117L * H * W), 256, 0, s>>>(oHead, cls_out + off * 9 * 13, 117, H, W, 13, 1);
    }

    // ---- ROI pool + recognition + NER ----
    long pooledN = 128L * 256 * POOL_H * POOL_W;
    fill_zero_kernel<<<g1(pooledN), 256, 0, s>>>(pooled, pooledN);
    roi_pool_kernel<<<dim3(64, 256), 128, 0, s>>>(p3, rois, pooled, 256, 100, 128);

    launch_conv(s, pooled, rec_c1, nullptr, nullptr, nullptr, rec_h,
                256, POOL_H, POOL_W, 256, POOL_H, 3, 1, 0, 1, 1,
                64, 256L * POOL_H * POOL_W, 256L * POOL_W);
    launch_conv(s, rec_h, rec_ow, nullptr, rec_ob, nullptr, rec_o,
                256, 1, POOL_W, 81, 1, 1, 1, 0, 0, 0,
                64, 256L * POOL_W, 81L * POOL_W);
    rec_logsoftmax_kernel<<<g1(64L * POOL_W), 256, 0, s>>>(rec_o, trans_out, 64, 81, POOL_W);

    launch_conv(s, pooled, ner_c1, nullptr, nullptr, nullptr, ner_h,
                256, POOL_H, POOL_W, 256, POOL_H, 3, 1, 0, 1, 1,
                128, 256L * POOL_H * POOL_W, 256L * POOL_W);
    ner_mean_fc_kernel<<<128, 256, 0, s>>>(ner_h, ner_w, ner_b, ner_out, 256, POOL_W, 13);
}
